// Layer_Attention_Dynamic_GATWithBias_48155173323386
// MI455X (gfx1250) — matrix operand order repaired and checked by reading, not compiled, not run on hardware
//
#include <hip/hip_runtime.h>

#define N 4096
#define F 512
#define R 64
#define H 8
#define EPS_BIAS 0.01f
#define ALPHA 0.2f

typedef __attribute__((ext_vector_type(2)))  float  v2f;
typedef __attribute__((ext_vector_type(8)))  float  v8f;
typedef __attribute__((ext_vector_type(4)))  int    v4i;
typedef __attribute__((ext_vector_type(16))) __bf16 v16bf;

#if __has_builtin(__builtin_amdgcn_global_load_async_to_lds_b128)
#define HAVE_ASYNC_LDS 1
#else
#define HAVE_ASYNC_LDS 0
#endif

// ---------------------------------------------------------------------------
// Pass 1: hidden[h][i][r] = sum_k node[i][k] * W[h][k][r]   (fp32 WMMA 16x16x4)
// Writes hidden (f32, [H][N][R]) for the score pass and hiddenT (bf16, [H][R][N])
// laid out so pass-3 B fragments are contiguous 32-byte loads.
// Block = 128 threads (4 waves); wave w handles r-block w. Grid = (N/16, H).
// ---------------------------------------------------------------------------
__global__ __launch_bounds__(128) void gat_hidden_kernel(
    const float* __restrict__ node, const float* __restrict__ W,
    float* __restrict__ hidden, __bf16* __restrict__ hiddenT)
{
    const int lane = threadIdx.x & 31;
    const int rblk = threadIdx.x >> 5;   // 0..3
    const int iblk = blockIdx.x;         // 0..255
    const int h    = blockIdx.y;         // 0..7
    const int i0   = iblk * 16;
    const int r0   = rblk * 16;
    const int mrow = lane & 15;
    const int half = lane >> 4;          // 0: lanes 0-15, 1: lanes 16-31

    // A frag (16x4 f32): lane holds node[i0+mrow][k0 + 2*half + {0,1}]
    const float* arow = node + (size_t)(i0 + mrow) * F + 2 * half;
    // B frag (4x16 f32): lane holds W[h][k0 + 2*half + {0,1}][r0 + mrow]
    const float* bcol = W + (size_t)h * F * R + (size_t)(2 * half) * R + (r0 + mrow);

    v8f c = {0.f, 0.f, 0.f, 0.f, 0.f, 0.f, 0.f, 0.f};
#pragma unroll 4
    for (int k0 = 0; k0 < F; k0 += 4) {
        v2f a = *(const v2f*)(arow + k0);
        v2f b;
        b.x = bcol[(size_t)k0 * R];
        b.y = bcol[(size_t)k0 * R + R];
        c = __builtin_amdgcn_wmma_f32_16x16x4_f32(false, a, false, b,
                                                  (short)0, c, false, false);
    }

#pragma unroll
    for (int v = 0; v < 8; ++v) {
        const int row = i0 + v + 8 * half;
        const int col = r0 + mrow;
        const float val = c[v];
        hidden [((size_t)h * N + row) * R + col] = val;
        hiddenT[((size_t)h * R + col) * N + row] = (__bf16)val;
    }
}

// ---------------------------------------------------------------------------
// Pass 2: a_src[h][i] = hidden[h][i][:] . sa[h][:R]
//         a_dst[h][i] = hidden[h][i][:] . sa[h][R:]
// Grid = (N/256, H), block = 256.
// ---------------------------------------------------------------------------
__global__ __launch_bounds__(256) void gat_scores_kernel(
    const float* __restrict__ hidden, const float* __restrict__ sa,
    float* __restrict__ a_src, float* __restrict__ a_dst)
{
    const int i = blockIdx.x * 256 + threadIdx.x;
    const int h = blockIdx.y;
    const float* hrow = hidden + ((size_t)h * N + i) * R;
    const float* sah  = sa + (size_t)h * (2 * R);
    float s = 0.f, d = 0.f;
#pragma unroll 8
    for (int r = 0; r < R; ++r) {
        const float hv = hrow[r];
        s += hv * sah[r];
        d += hv * sah[R + r];
    }
    a_src[(size_t)h * N + i] = s;
    a_dst[(size_t)h * N + i] = d;
}

// ---------------------------------------------------------------------------
// Pass 3: fused score -> exp -> (bf16 WMMA) aggregation, with the softmax
// denominator computed by a 5th WMMA against an all-ones B fragment (lands in
// the same C-layout slots used to scale the output -> zero shuffles).
// Block = 256 (8 waves, wave = head), grid = N/16 (one 16-row i-block each).
// All 8 waves stream the same adj/bias rows -> WGP$ reuse (1 HBM pass).
// a_dst is staged to LDS once via async global->LDS copies.
// ---------------------------------------------------------------------------
__device__ __forceinline__ float gat_p(float asrc, float adj, float bias, float adst)
{
    const float t = asrc + adst + EPS_BIAS * bias;
    const float l = fmaxf(t, ALPHA * t);          // LeakyReLU
    return adj * __expf(l);                       // masked -> exactly 0
}

__global__ __launch_bounds__(256) void gat_attention_kernel(
    const float* __restrict__ adj, const float* __restrict__ bias,
    const float* __restrict__ a_src, const float* __restrict__ a_dst,
    const __bf16* __restrict__ hiddenT, float* __restrict__ out)
{
    const int lane = threadIdx.x & 31;
    const int h    = threadIdx.x >> 5;   // wave id = head
    const int i0   = blockIdx.x * 16;
    const int mrow = lane & 15;
    const int half = lane >> 4;

    const int i = i0 + mrow;
    const float asrc = a_src[(size_t)h * N + i];
    const float* adjrow  = adj  + (size_t)i * N;
    const float* biasrow = bias + (size_t)i * N;
    const float* adh     = a_dst + (size_t)h * N;

    // Stage this wave's a_dst[h][0..N) into LDS (wave-private region, no barrier).
    __shared__ __align__(16) float s_adst[H * N];
    float* ldsl = &s_adst[h * N];
#if HAVE_ASYNC_LDS
#pragma unroll 4
    for (int t = lane * 4; t < N; t += 32 * 4) {
        __builtin_amdgcn_global_load_async_to_lds_b128(
            (v4i*)(void*)(adh + t),
            (v4i*)(void*)(ldsl + t),
            0, 0);
    }
    asm volatile("s_wait_asynccnt 0" ::: "memory");
#else
    for (int t = lane * 4; t < N; t += 32 * 4) {
        *(float4*)(ldsl + t) = *(const float4*)(adh + t);
    }
#endif

    // bf16 A frag (16x32): lane<16 holds K = 0..15, lane>=16 K = 16..31
    const int offA = half * 16;
    // bf16 B frag (32x16): lanes 0-15 hold K rows 0..15, lanes 16-31 hold 16..31; N col = mrow
    const int offB = half * 16;
    const __bf16* hT0 = hiddenT + ((size_t)h * R +  0 + mrow) * N + offB;
    const __bf16* hT1 = hiddenT + ((size_t)h * R + 16 + mrow) * N + offB;
    const __bf16* hT2 = hiddenT + ((size_t)h * R + 32 + mrow) * N + offB;
    const __bf16* hT3 = hiddenT + ((size_t)h * R + 48 + mrow) * N + offB;

    v16bf Bones;
#pragma unroll
    for (int e = 0; e < 16; ++e) Bones[e] = (__bf16)1.0f;

    v8f c0 = {0.f,0.f,0.f,0.f,0.f,0.f,0.f,0.f};
    v8f c1 = c0, c2 = c0, c3 = c0;
    v8f csum = c0;   // per-row softmax denominator, accumulated on the XDL pipe

    for (int j0 = 0; j0 < N; j0 += 32) {
        // prefetch next tile of the streamed 64MB matrices (global_prefetch_b8)
        __builtin_prefetch(adjrow  + j0 + offA + 128, 0, 0);
        __builtin_prefetch(biasrow + j0 + offA + 128, 0, 0);

        const float4 aj0 = *(const float4*)(adjrow  + j0 + offA);
        const float4 aj1 = *(const float4*)(adjrow  + j0 + offA + 4);
        const float4 aj2 = *(const float4*)(adjrow  + j0 + offA + 8);
        const float4 aj3 = *(const float4*)(adjrow  + j0 + offA + 12);
        const float4 bj0 = *(const float4*)(biasrow + j0 + offA);
        const float4 bj1 = *(const float4*)(biasrow + j0 + offA + 4);
        const float4 bj2 = *(const float4*)(biasrow + j0 + offA + 8);
        const float4 bj3 = *(const float4*)(biasrow + j0 + offA + 12);
        const float4 dj0 = *(const float4*)(ldsl + j0 + offA);
        const float4 dj1 = *(const float4*)(ldsl + j0 + offA + 4);
        const float4 dj2 = *(const float4*)(ldsl + j0 + offA + 8);
        const float4 dj3 = *(const float4*)(ldsl + j0 + offA + 12);

        v16bf Af;
#define GAT_SC(e, A4, B4, D4, comp)                                   \
        { const float p = gat_p(asrc, A4.comp, B4.comp, D4.comp);     \
          Af[e] = (__bf16)p; }
        GAT_SC(0,  aj0, bj0, dj0, x)  GAT_SC(1,  aj0, bj0, dj0, y)
        GAT_SC(2,  aj0, bj0, dj0, z)  GAT_SC(3,  aj0, bj0, dj0, w)
        GAT_SC(4,  aj1, bj1, dj1, x)  GAT_SC(5,  aj1, bj1, dj1, y)
        GAT_SC(6,  aj1, bj1, dj1, z)  GAT_SC(7,  aj1, bj1, dj1, w)
        GAT_SC(8,  aj2, bj2, dj2, x)  GAT_SC(9,  aj2, bj2, dj2, y)
        GAT_SC(10, aj2, bj2, dj2, z)  GAT_SC(11, aj2, bj2, dj2, w)
        GAT_SC(12, aj3, bj3, dj3, x)  GAT_SC(13, aj3, bj3, dj3, y)
        GAT_SC(14, aj3, bj3, dj3, z)  GAT_SC(15, aj3, bj3, dj3, w)
#undef GAT_SC

        const v16bf B0 = *(const v16bf*)(hT0 + j0);
        const v16bf B1 = *(const v16bf*)(hT1 + j0);
        const v16bf B2 = *(const v16bf*)(hT2 + j0);
        const v16bf B3 = *(const v16bf*)(hT3 + j0);

        c0   = __builtin_amdgcn_wmma_f32_16x16x32_bf16(false, Af, false, B0,    (short)0, c0,   false, false);
        c1   = __builtin_amdgcn_wmma_f32_16x16x32_bf16(false, Af, false, B1,    (short)0, c1,   false, false);
        c2   = __builtin_amdgcn_wmma_f32_16x16x32_bf16(false, Af, false, B2,    (short)0, c2,   false, false);
        c3   = __builtin_amdgcn_wmma_f32_16x16x32_bf16(false, Af, false, B3,    (short)0, c3,   false, false);
        csum = __builtin_amdgcn_wmma_f32_16x16x32_bf16(false, Af, false, Bones, (short)0, csum, false, false);
    }

    // csum[v] (any lane in a half) = softmax denominator for row v + 8*half:
    // exactly the row owned by C-slot (v, half) -> scale in place, no shuffles.
    // out[i, h*R + r], row = i0 + v + 8*half, col r = 16*t + mrow
#pragma unroll
    for (int v = 0; v < 8; ++v) {
        const float sc = 1.0f / csum[v];   // self-loop guarantees > 0
        float* orow = out + (size_t)(i0 + v + 8 * half) * (H * R) + (size_t)h * R + mrow;
        orow[ 0] = c0[v] * sc;
        orow[16] = c1[v] * sc;
        orow[32] = c2[v] * sc;
        orow[48] = c3[v] * sc;
    }
}

// ---------------------------------------------------------------------------
extern "C" void kernel_launch(void* const* d_in, const int* in_sizes, int n_in,
                              void* d_out, int out_size, void* d_ws, size_t ws_size,
                              hipStream_t stream)
{
    const float* node = (const float*)d_in[0];   // [N, F]
    const float* adj  = (const float*)d_in[1];   // [N, N]
    const float* W    = (const float*)d_in[2];   // [H, F, R]
    const float* sa   = (const float*)d_in[3];   // [H, 2R]
    const float* bias = (const float*)d_in[4];   // [N, N]
    float* out = (float*)d_out;                  // [N, H*R]

    char* ws = (char*)d_ws;
    float*  hidden  = (float*)ws;                                   // H*N*R f32 (8 MB)
    __bf16* hiddenT = (__bf16*)(ws + (size_t)H * N * R * 4);        // H*R*N bf16 (4 MB)
    float*  a_src   = (float*)(ws + (size_t)H * N * R * 4 + (size_t)H * R * N * 2);
    float*  a_dst   = a_src + (size_t)H * N;

    gat_hidden_kernel<<<dim3(N / 16, H), 128, 0, stream>>>(node, W, hidden, hiddenT);
    gat_scores_kernel<<<dim3(N / 256, H), 256, 0, stream>>>(hidden, sa, a_src, a_dst);
    gat_attention_kernel<<<dim3(N / 16), 256, 0, stream>>>(adj, bias, a_src, a_dst, hiddenT, out);

    (void)in_sizes; (void)n_in; (void)out_size; (void)ws_size;
}
